// VITS_24842090840727
// MI455X (gfx1250) — matrix operand order repaired and checked by reading, not compiled, not run on hardware
//
#include <hip/hip_runtime.h>

typedef _Float16 f16;
typedef __attribute__((ext_vector_type(16))) _Float16 v16h;
typedef __attribute__((ext_vector_type(8)))  _Float16 v8h;
typedef __attribute__((ext_vector_type(8)))  float    v8f;

#define Bn 32
#define Tn 2048
#define Hn 192
#define Ln 16
#define Kn 5
#define NT 32           // time-tile per workgroup
#define HALO 2

__device__ __forceinline__ v8f wmma_f16(v16h a, v16h b, v8f c) {
  return __builtin_amdgcn_wmma_f32_16x16x32_f16(false, a, false, b, (short)0, c, false, false);
}

__device__ __forceinline__ void wait_asynccnt0() {
#if __has_builtin(__builtin_amdgcn_s_wait_asynccnt)
  __builtin_amdgcn_s_wait_asynccnt(0);
#else
  asm volatile("s_wait_asynccnt 0x0" ::: "memory");
#endif
}

// Async 16B copy global -> LDS (CDNA5 GLOBAL_LOAD_ASYNC_TO_LDS_B128, ASYNCcnt-tracked).
__device__ __forceinline__ void async_copy_b128(unsigned lds_off, unsigned long long gaddr) {
  asm volatile("global_load_async_to_lds_b128 %0, %1, off"
               :: "v"(lds_off), "v"(gaddr) : "memory");
}

// A fragment (16x32 f16, M x K), row-major with row stride Hn (=192) f16.
// ISA layout: lane m=lane&15; lanes<16 hold K = {0..7, 16..23}, lanes>=16 hold K = {8..15, 24..31}.
__device__ __forceinline__ v16h load_a(const f16* __restrict__ base, int lane) {
  const int m  = lane & 15;
  const int kh = (lane >> 4) << 3;            // 0 or 8
  const f16* p = base + m * Hn + kh;
  v8h lo = *(const v8h*)(p);                  // K = kh .. kh+7
  v8h hi = *(const v8h*)(p + 16);             // K = 16+kh .. 16+kh+7
  v16h r;
#pragma unroll
  for (int i = 0; i < 8; ++i) { r[i] = lo[i]; r[i + 8] = hi[i]; }
  return r;
}

// B fragment (32x16 f16, K x N) from LDS laid out [time][chan] (chan contiguous, row stride Hn).
// ISA layout: column n=lane&15; lanes<16 hold K = {0..7, 16..23}, lanes>=16 hold K = {8..15, 24..31}.
__device__ __forceinline__ v16h load_b(const f16* __restrict__ p, int lane) {
  const int n  = lane & 15;
  const int ko = (lane >> 4) << 3;            // 0 or 8
  const f16* q = p + n * Hn + ko;
  v8h lo = *(const v8h*)(q);
  v8h hi = *(const v8h*)(q + 16);
  v16h r;
#pragma unroll
  for (int i = 0; i < 8; ++i) { r[i] = lo[i]; r[i + 8] = hi[i]; }
  return r;
}

// ---------------- weight conversion (once per launch, trivial cost) ----------------
// wn_in_W  f32 [L][2H][H][K]  -> f16 [L][K][2H][H]
// wn_rs_W  f32 [L][2H][H][1]  -> f16 [L][2H][H]
// post_W   f32 [2C][H][1]     -> f16 [2C][H]
__global__ void convert_weights(const float* __restrict__ wIn, const float* __restrict__ wRs,
                                const float* __restrict__ wPost,
                                f16* __restrict__ wInH, f16* __restrict__ wRsH,
                                f16* __restrict__ wPostH) {
  const long long N1 = (long long)Ln * Kn * 384 * Hn;
  const long long N2 = (long long)Ln * 384 * Hn;
  const long long N3 = (long long)384 * Hn;
  long long id = (long long)blockIdx.x * blockDim.x + threadIdx.x;
  if (id < N1) {
    int ic = (int)(id % Hn);
    int oc = (int)((id / Hn) % 384);
    int k  = (int)((id / ((long long)Hn * 384)) % Kn);
    int l  = (int)(id / ((long long)Hn * 384 * Kn));
    wInH[id] = (f16)wIn[(((long long)l * 384 + oc) * Hn + ic) * Kn + k];
  } else if (id < N1 + N2) {
    long long j = id - N1;
    wRsH[j] = (f16)wRs[j];
  } else if (id < N1 + N2 + N3) {
    long long j = id - N1 - N2;
    wPostH[j] = (f16)wPost[j];
  }
}

// ---------------- pre 1x1 conv (80 -> 192), masked, f16 output [B][T][H] ----------------
__global__ __launch_bounds__(256) void pre_kernel(const float* __restrict__ x,
                                                  const float* __restrict__ preW,
                                                  const float* __restrict__ preB,
                                                  const int* __restrict__ xlen,
                                                  f16* __restrict__ h0) {
  long long id = (long long)blockIdx.x * blockDim.x + threadIdx.x;
  if (id >= (long long)Bn * Tn * Hn) return;
  int c = (int)(id % Hn);
  int t = (int)((id / Hn) % Tn);
  int b = (int)(id / ((long long)Hn * Tn));
  float acc = preB[c];
  const float* xp = x + ((long long)b * 80) * Tn + t;
  const float* wp = preW + c * 80;
#pragma unroll 8
  for (int ic = 0; ic < 80; ++ic) acc += wp[ic] * xp[(long long)ic * Tn];
  if (t >= xlen[b]) acc = 0.0f;
  h0[id] = (f16)acc;
}

// ---------------- fused WN layer: K=5 conv -> tanh*sigmoid -> 1x1 conv -> h/out ----------------
__global__ __launch_bounds__(256) void wn_layer_kernel(
    const f16* __restrict__ h_in, f16* __restrict__ h_out, float* __restrict__ outAcc,
    const f16* __restrict__ wInH,   // [5][384][192] for this layer
    const float* __restrict__ bIn,  // [384]
    const f16* __restrict__ wRsH,   // [384][192]
    const float* __restrict__ bRs,  // [384]
    const int* __restrict__ xlen, int last) {
  __shared__ alignas(16) f16 sh_h[(NT + 2 * HALO) * Hn];   // 13824 B
  __shared__ alignas(16) f16 sh_xin[384 * 33];             // 25344 B
  __shared__ alignas(16) f16 sh_acts[NT * Hn];             // 12288 B

  const int tid  = threadIdx.x;
  const int lane = tid & 31;
  const int wave = tid >> 5;
  const int nblk = Tn / NT;
  const int b  = blockIdx.x / nblk;
  const int t0 = (blockIdx.x % nblk) * NT;
  const int len = xlen[b];

  // stage h tile with +-2 halo into LDS: async DMA for in-range rows (no VGPR round-trip),
  // plain zero stores for out-of-range halo rows (only at sequence boundaries).
  for (int i = tid; i < (NT + 2 * HALO) * (Hn / 8); i += 256) {
    int row = i / (Hn / 8);
    int col = (i % (Hn / 8)) * 8;
    int t = t0 - HALO + row;
    f16* dst = sh_h + row * Hn + col;
    if (t >= 0 && t < Tn) {
      async_copy_b128((unsigned)(uintptr_t)dst,
                      (unsigned long long)(h_in + ((size_t)b * Tn + t) * Hn + col));
    } else {
      *(uint4*)dst = make_uint4(0u, 0u, 0u, 0u);
    }
  }
  wait_asynccnt0();
  __syncthreads();

  const int mbase = wave * 48;
  const int mhi   = (lane >> 4) << 3;
  const int ncol  = lane & 15;

  // ---- x_in = conv5(h) + bIn ----
  v8f acc[6];
#pragma unroll
  for (int mt = 0; mt < 3; ++mt) {
    v8f a;
#pragma unroll
    for (int r = 0; r < 8; ++r) a[r] = bIn[mbase + mt * 16 + mhi + r];
    acc[mt * 2 + 0] = a;
    acc[mt * 2 + 1] = a;
  }
  for (int tap = 0; tap < Kn; ++tap) {
#pragma unroll
    for (int kk = 0; kk < Hn / 32; ++kk) {
      const f16* wp = wInH + ((size_t)(tap * 384 + mbase)) * Hn + kk * 32;
      v16h a0 = load_a(wp, lane);
      v16h a1 = load_a(wp + 16 * Hn, lane);
      v16h a2 = load_a(wp + 32 * Hn, lane);
      const f16* hp = sh_h + tap * Hn + kk * 32;   // local time = n + tap (halo origin t0-2)
      v16h b0 = load_b(hp, lane);
      v16h b1 = load_b(hp + 16 * Hn, lane);
      acc[0] = wmma_f16(a0, b0, acc[0]);  acc[1] = wmma_f16(a0, b1, acc[1]);
      acc[2] = wmma_f16(a1, b0, acc[2]);  acc[3] = wmma_f16(a1, b1, acc[3]);
      acc[4] = wmma_f16(a2, b0, acc[4]);  acc[5] = wmma_f16(a2, b1, acc[5]);
    }
  }
  // spill x_in to LDS (f16) for the cross-wave gated activation
#pragma unroll
  for (int mt = 0; mt < 3; ++mt)
#pragma unroll
    for (int nt = 0; nt < 2; ++nt) {
      v8f& A = acc[mt * 2 + nt];
#pragma unroll
      for (int r = 0; r < 8; ++r)
        sh_xin[(mbase + mt * 16 + mhi + r) * 33 + nt * 16 + ncol] = (f16)A[r];
    }
  __syncthreads();

  // ---- acts = tanh(x_in[:H]) * sigmoid(x_in[H:]) ----
  for (int i = tid; i < NT * Hn; i += 256) {
    int n = i / Hn, c = i % Hn;
    float xa = (float)sh_xin[c * 33 + n];
    float xb = (float)sh_xin[(c + Hn) * 33 + n];
    float th = tanhf(xa);
    float sg = 1.0f / (1.0f + __expf(-xb));
    sh_acts[n * Hn + c] = (f16)(th * sg);
  }
  __syncthreads();

  // ---- rs = W_rs @ acts + bRs ----
#pragma unroll
  for (int mt = 0; mt < 3; ++mt) {
    v8f a;
#pragma unroll
    for (int r = 0; r < 8; ++r) a[r] = bRs[mbase + mt * 16 + mhi + r];
    acc[mt * 2 + 0] = a;
    acc[mt * 2 + 1] = a;
  }
#pragma unroll
  for (int kk = 0; kk < Hn / 32; ++kk) {
    const f16* wp = wRsH + (size_t)mbase * Hn + kk * 32;
    v16h a0 = load_a(wp, lane);
    v16h a1 = load_a(wp + 16 * Hn, lane);
    v16h a2 = load_a(wp + 32 * Hn, lane);
    const f16* ap = sh_acts + kk * 32;
    v16h b0 = load_b(ap, lane);
    v16h b1 = load_b(ap + 16 * Hn, lane);
    acc[0] = wmma_f16(a0, b0, acc[0]);  acc[1] = wmma_f16(a0, b1, acc[1]);
    acc[2] = wmma_f16(a1, b0, acc[2]);  acc[3] = wmma_f16(a1, b1, acc[3]);
    acc[4] = wmma_f16(a2, b0, acc[4]);  acc[5] = wmma_f16(a2, b1, acc[5]);
  }

  // ---- h/out update ----
#pragma unroll
  for (int mt = 0; mt < 3; ++mt)
#pragma unroll
    for (int nt = 0; nt < 2; ++nt) {
      v8f& A = acc[mt * 2 + nt];
#pragma unroll
      for (int r = 0; r < 8; ++r) {
        int m = mbase + mt * 16 + mhi + r;
        int n = nt * 16 + ncol;
        int t = t0 + n;
        size_t gi = ((size_t)b * Tn + t) * Hn;
        if (m < Hn) {
          if (!last) {  // h = (h + rs[:H]) * mask
            float hv = (float)sh_h[(n + HALO) * Hn + m];
            float nv = (hv + A[r]) * ((t < len) ? 1.0f : 0.0f);
            h_out[gi + m] = (f16)nv;
          } else {      // out += rs[:H]
            outAcc[gi + m] += A[r];
          }
        } else if (!last) {  // out += rs[H:]
          outAcc[gi + (m - Hn)] += A[r];
        }
      }
    }
}

// ---------------- post: stats = post_W @ out + post_b ; z = mu + eps*exp(log_s) ----------------
__global__ __launch_bounds__(256) void post_kernel(const float* __restrict__ outAcc,
                                                   const f16* __restrict__ wPostH,
                                                   const float* __restrict__ postB,
                                                   const float* __restrict__ eps,
                                                   const int* __restrict__ xlen,
                                                   float* __restrict__ z) {
  __shared__ alignas(16) f16   sh_out[NT * Hn];   // 12288 B
  __shared__ alignas(16) float sh_st[384 * 33];   // 50688 B

  const int tid  = threadIdx.x;
  const int lane = tid & 31;
  const int wave = tid >> 5;
  const int nblk = Tn / NT;
  const int b  = blockIdx.x / nblk;
  const int t0 = (blockIdx.x % nblk) * NT;
  const int len = xlen[b];

  for (int i = tid; i < NT * Hn; i += 256) {
    int n = i / Hn, c = i % Hn;
    sh_out[n * Hn + c] = (f16)outAcc[((size_t)b * Tn + t0 + n) * Hn + c];
  }
  __syncthreads();

  const int mbase = wave * 48;
  const int mhi   = (lane >> 4) << 3;
  const int ncol  = lane & 15;

  v8f acc[6];
#pragma unroll
  for (int mt = 0; mt < 3; ++mt) {
    v8f a;
#pragma unroll
    for (int r = 0; r < 8; ++r) a[r] = postB[mbase + mt * 16 + mhi + r];
    acc[mt * 2 + 0] = a;
    acc[mt * 2 + 1] = a;
  }
#pragma unroll
  for (int kk = 0; kk < Hn / 32; ++kk) {
    const f16* wp = wPostH + (size_t)mbase * Hn + kk * 32;
    v16h a0 = load_a(wp, lane);
    v16h a1 = load_a(wp + 16 * Hn, lane);
    v16h a2 = load_a(wp + 32 * Hn, lane);
    const f16* op = sh_out + kk * 32;
    v16h b0 = load_b(op, lane);
    v16h b1 = load_b(op + 16 * Hn, lane);
    acc[0] = wmma_f16(a0, b0, acc[0]);  acc[1] = wmma_f16(a0, b1, acc[1]);
    acc[2] = wmma_f16(a1, b0, acc[2]);  acc[3] = wmma_f16(a1, b1, acc[3]);
    acc[4] = wmma_f16(a2, b0, acc[4]);  acc[5] = wmma_f16(a2, b1, acc[5]);
  }
#pragma unroll
  for (int mt = 0; mt < 3; ++mt)
#pragma unroll
    for (int nt = 0; nt < 2; ++nt) {
      v8f& A = acc[mt * 2 + nt];
#pragma unroll
      for (int r = 0; r < 8; ++r)
        sh_st[(mbase + mt * 16 + mhi + r) * 33 + nt * 16 + ncol] = A[r];
    }
  __syncthreads();

  for (int i = tid; i < Hn * NT; i += 256) {
    int c = i / NT, n = i % NT;
    int t = t0 + n;
    float mu = sh_st[c * 33 + n];
    float ls = sh_st[(c + Hn) * 33 + n];
    size_t ei = ((size_t)b * Hn + c) * Tn + t;
    float e = eps[ei];
    z[ei] = (t < len) ? (mu + e * __expf(ls)) : e;
  }
}

extern "C" void kernel_launch(void* const* d_in, const int* in_sizes, int n_in,
                              void* d_out, int out_size, void* d_ws, size_t ws_size,
                              hipStream_t stream) {
  const float* x       = (const float*)d_in[0];
  const int*   xlen    = (const int*)d_in[1];
  const float* eps     = (const float*)d_in[2];
  const float* pre_W   = (const float*)d_in[3];
  const float* pre_b   = (const float*)d_in[4];
  const float* wn_in_W = (const float*)d_in[5];
  const float* wn_in_b = (const float*)d_in[6];
  const float* wn_rs_W = (const float*)d_in[7];
  const float* wn_rs_b = (const float*)d_in[8];
  const float* post_W  = (const float*)d_in[9];
  const float* post_b  = (const float*)d_in[10];

  char* ws = (char*)d_ws;
  size_t off = 0;
  auto alloc = [&](size_t bytes) {
    void* p = ws + off;
    off = (off + bytes + 255) & ~(size_t)255;
    return p;
  };
  const size_t nH = (size_t)Bn * Tn * Hn;
  f16*   hA     = (f16*)alloc(nH * sizeof(f16));
  f16*   hB     = (f16*)alloc(nH * sizeof(f16));
  float* outAcc = (float*)alloc(nH * sizeof(float));
  f16*   wInH   = (f16*)alloc((size_t)Ln * Kn * 384 * Hn * sizeof(f16));
  f16*   wRsH   = (f16*)alloc((size_t)Ln * 384 * Hn * sizeof(f16));
  f16*   wPostH = (f16*)alloc((size_t)384 * Hn * sizeof(f16));

  hipMemsetAsync(outAcc, 0, nH * sizeof(float), stream);

  {
    long long nconv = (long long)Ln * Kn * 384 * Hn + (long long)Ln * 384 * Hn + (long long)384 * Hn;
    int blocks = (int)((nconv + 255) / 256);
    convert_weights<<<blocks, 256, 0, stream>>>(wn_in_W, wn_rs_W, post_W, wInH, wRsH, wPostH);
  }
  {
    long long npre = (long long)Bn * Tn * Hn;
    int blocks = (int)((npre + 255) / 256);
    pre_kernel<<<blocks, 256, 0, stream>>>(x, pre_W, pre_b, xlen, hA);
  }

  const int grid = Bn * (Tn / NT);
  for (int l = 0; l < Ln; ++l) {
    const f16* hin  = (l & 1) ? hB : hA;
    f16*       hout = (l & 1) ? hA : hB;
    wn_layer_kernel<<<grid, 256, 0, stream>>>(
        hin, hout, outAcc,
        wInH + (size_t)l * Kn * 384 * Hn, wn_in_b + (size_t)l * 384,
        wRsH + (size_t)l * 384 * Hn,      wn_rs_b + (size_t)l * 384,
        xlen, (l == Ln - 1) ? 1 : 0);
  }

  post_kernel<<<grid, 256, 0, stream>>>(outAcc, wPostH, post_b, eps, xlen, (float*)d_out);
}
